// Network_13151189860652
// MI455X (gfx1250) — matrix operand order repaired and checked by reading, not compiled, not run on hardware
//
#include <hip/hip_runtime.h>

typedef __attribute__((ext_vector_type(16))) _Float16 v16h;
typedef __attribute__((ext_vector_type(8)))  _Float16 v8h;
typedef __attribute__((ext_vector_type(2)))  _Float16 v2h;
typedef __attribute__((ext_vector_type(8)))  float    v8f;

#define B_    64
#define T_    500
#define FIN   1250
#define KPAD  1280
#define FH    512
#define FOUT  2
#define M_    (B_ * T_)   // 32000 rows, divisible by 16
#define NITER (KPAD / 32) // 40 K-steps; steps 0..38 are full (k0+32 <= 1250)

// SLAYER constants
#define DECAY_SR  0.90483741803595957f   // exp(-1/10)
#define DECAY_REF 0.36787944117144233f   // exp(-1)
#define THETA     10.0f
#define REF_DROP  20.0f                  // SCALE_REF * THETA

#define LROW 40   // LDS row stride in halves (80B) -> conflict-free b128 reads

// ---------------------------------------------------------------------------
// Kernel 0: convert W1 (512 x 1250 f32) -> W1h (512 x 1280 f16, zero padded K)
// ---------------------------------------------------------------------------
__global__ void w1_to_f16(const float* __restrict__ W1, _Float16* __restrict__ W1h) {
    int idx = blockIdx.x * blockDim.x + threadIdx.x;
    if (idx >= FH * KPAD) return;
    int n = idx / KPAD;
    int k = idx - n * KPAD;
    float v = (k < FIN) ? W1[(size_t)n * FIN + k] : 0.0f;
    W1h[idx] = (_Float16)v;
}

// ---------------------------------------------------------------------------
// LDS staging helpers for the 16x32 x-tile (f32 -> f16).
// 256 threads: each loads one aligned float2 (row stride 5000B => 8B aligned),
// packs to 2 halves (v_cvt_pk_f16_f32), one ds_store_b32. No predicates on
// the fast path; the bounds-checked tail runs exactly once (last K-step).
// ---------------------------------------------------------------------------
__device__ __forceinline__ void stage_full(const float* __restrict__ x, int m0,
                                           int k0, _Float16* __restrict__ buf,
                                           int tid) {
    int r  = tid >> 4;             // 0..15
    int kp = (tid & 15) << 1;      // 0,2,..,30
    const float2 v = *(const float2*)&x[(size_t)(m0 + r) * FIN + k0 + kp];
    v2h h;
    h[0] = (_Float16)v.x;
    h[1] = (_Float16)v.y;
    *(v2h*)&buf[r * LROW + kp] = h;
}

__device__ __forceinline__ void stage_tail(const float* __restrict__ x, int m0,
                                           int k0, _Float16* __restrict__ buf,
                                           int tid) {
    int r  = tid >> 4;
    int kp = (tid & 15) << 1;
    v2h h;
    #pragma unroll
    for (int j = 0; j < 2; ++j) {
        int kg = k0 + kp + j;
        float v = (kg < FIN) ? x[(size_t)(m0 + r) * FIN + kg] : 0.0f;
        h[j] = (_Float16)v;
    }
    *(v2h*)&buf[r * LROW + kp] = h;
}

// ---------------------------------------------------------------------------
// Kernel 1: u1[m, n] = sum_k x[m,k] * W1[n,k]   (M=32000, N=512, K=1250)
// Block: 256 threads (8 waves). M-tile = 16 rows, full N=512 per block.
// Wave w owns N range [w*64, w*64+64) as 4 x (16x16) WMMA subtiles.
// Double-buffered LDS, one barrier per K-step. All 4 B fragments are
// preloaded into distinct registers so the compiler can use partial
// s_wait_loadcnt and overlap WMMAs with in-flight B loads.
// ---------------------------------------------------------------------------
__launch_bounds__(256)
__global__ void gemm1_wmma(const float* __restrict__ x,
                           const _Float16* __restrict__ W1h,
                           float* __restrict__ u1) {
    __shared__ __align__(32) _Float16 lds[2][16 * LROW];

    const int m0     = blockIdx.x * 16;
    const int tid    = threadIdx.x;
    const int wave   = tid >> 5;
    const int lane   = tid & 31;
    const int row    = lane & 15;   // matrix row (A) / column (B,D)
    const int hiHalf = lane >> 4;   // 0: lanes 0-15, 1: lanes 16-31
    const int kbA    = hiHalf * 8;
    const int kbB    = hiHalf * 16;

    // per-lane base for B fragments: column n = wave*64 + s*16 + row
    const _Float16* __restrict__ bbase =
        W1h + (size_t)(wave * 64 + row) * KPAD + kbB;

    v8f acc[4] = {v8f{}, v8f{}, v8f{}, v8f{}};

    stage_full(x, m0, 0, lds[0], tid);
    __syncthreads();

    for (int it = 0; it < NITER; ++it) {
        const int k0 = it * 32;
        _Float16* cur = lds[it & 1];
        _Float16* nxt = lds[(it & 1) ^ 1];

        // prefetch next x-tile into the other buffer (no barrier needed yet)
        if (it + 1 < NITER) {
            const int kn = k0 + 32;
            if (kn + 32 <= FIN) stage_full(x, m0, kn, nxt, tid);
            else                stage_tail(x, m0, kn, nxt, tid);
        }

        // Issue all 4 B-fragment loads first (distinct registers -> one big
        // clause, partial waits before each WMMA).
        v16h b[4];
        #pragma unroll
        for (int s = 0; s < 4; ++s)
            b[s] = *(const v16h*)(bbase + (size_t)s * 16 * KPAD + k0);

        // A fragment (16x32 f16): lane half h -> K {16h..16h+15} (contiguous)
        v8h alo = *(const v8h*)&cur[row * LROW + kbB];
        v8h ahi = *(const v8h*)&cur[row * LROW + kbB + 8];
        v16h a;
        #pragma unroll
        for (int i = 0; i < 8; ++i) { a[i] = alo[i]; a[i + 8] = ahi[i]; }

        #pragma unroll
        for (int s = 0; s < 4; ++s) {
            acc[s] = __builtin_amdgcn_wmma_f32_16x16x32_f16(
                /*neg_a=*/false, a, /*neg_b=*/false, b[s],
                /*c_mod=*/(short)0, acc[s],
                /*reuse_a=*/false, /*reuse_b=*/false);
        }
        __syncthreads();   // publishes nxt, retires reads of cur
    }

    // D layout: element e holds m = m0 + e + 8*hiHalf, n = lane&15
    #pragma unroll
    for (int s = 0; s < 4; ++s) {
        int n = wave * 64 + s * 16 + row;
        #pragma unroll
        for (int e = 0; e < 8; ++e) {
            int m = m0 + e + 8 * hiHalf;
            u1[(size_t)m * FH + n] = acc[s][e];
        }
    }
}

// ---------------------------------------------------------------------------
// Kernel 2: layer-1 temporal PSP + refractory spike scan.
// One thread per (b, h) chain; coalesced over h (stride-1).
// ---------------------------------------------------------------------------
__global__ void scan1(const float* __restrict__ u1, _Float16* __restrict__ s1) {
    int idx = blockIdx.x * blockDim.x + threadIdx.x;   // 0 .. B_*FH-1
    int b = idx >> 9;          // / 512
    int h = idx & (FH - 1);
    size_t base = (size_t)b * T_ * FH + h;
    float p = 0.0f, r = 0.0f;
    for (int t = 0; t < T_; ++t) {
        float u = u1[base + (size_t)t * FH];
        p = DECAY_SR * p + u;             // PSP filter
        float v = p + r;                  // membrane + refractory
        float s = (v >= THETA) ? 1.0f : 0.0f;
        r = DECAY_REF * (r - REF_DROP * s);
        s1[base + (size_t)t * FH] = (_Float16)s;
    }
}

// ---------------------------------------------------------------------------
// Kernel 3: u2[m, o] = sum_h s1[m,h] * W2[o,h]   (N=2: per-thread dual dot)
// ---------------------------------------------------------------------------
__global__ void layer2_mm(const _Float16* __restrict__ s1,
                          const float* __restrict__ W2,
                          float* __restrict__ u2) {
    int m = blockIdx.x * blockDim.x + threadIdx.x;
    if (m >= M_) return;
    const _Float16* sp = s1 + (size_t)m * FH;
    float a0 = 0.0f, a1 = 0.0f;
    #pragma unroll 8
    for (int h = 0; h < FH; ++h) {
        float sv = (float)sp[h];
        a0 += sv * W2[h];
        a1 += sv * W2[FH + h];
    }
    u2[m * 2 + 0] = a0;
    u2[m * 2 + 1] = a1;
}

// ---------------------------------------------------------------------------
// Kernel 4: layer-2 temporal scan (128 chains), f32 output.
// ---------------------------------------------------------------------------
__global__ void scan2(const float* __restrict__ u2, float* __restrict__ out) {
    int idx = threadIdx.x;            // 0 .. 127
    int b = idx >> 1;
    int o = idx & 1;
    size_t base = (size_t)b * T_ * FOUT + o;
    float p = 0.0f, r = 0.0f;
    for (int t = 0; t < T_; ++t) {
        float u = u2[base + (size_t)t * FOUT];
        p = DECAY_SR * p + u;
        float v = p + r;
        float s = (v >= THETA) ? 1.0f : 0.0f;
        r = DECAY_REF * (r - REF_DROP * s);
        out[base + (size_t)t * FOUT] = s;
    }
}

// ---------------------------------------------------------------------------
// Workspace layout (bytes):
//   W1h : FH*KPAD*2            =  1,310,720
//   u1  : M_*FH*4              = 65,536,000
//   s1  : M_*FH*2              = 32,768,000
//   u2  : M_*FOUT*4            =    256,000
// total ~95.2 MB, fully overwritten every call (no cross-call state).
// ---------------------------------------------------------------------------
extern "C" void kernel_launch(void* const* d_in, const int* in_sizes, int n_in,
                              void* d_out, int out_size, void* d_ws, size_t ws_size,
                              hipStream_t stream) {
    const float* x  = (const float*)d_in[0];   // (64, 500, 1250)
    const float* W1 = (const float*)d_in[1];   // (512, 1250)
    const float* W2 = (const float*)d_in[2];   // (2, 512)

    char* ws = (char*)d_ws;
    const size_t OFF_W1H = 0;
    const size_t OFF_U1  = OFF_W1H + (size_t)FH * KPAD * sizeof(_Float16);
    const size_t OFF_S1  = OFF_U1  + (size_t)M_ * FH * sizeof(float);
    const size_t OFF_U2  = OFF_S1  + (size_t)M_ * FH * sizeof(_Float16);

    _Float16* W1h = (_Float16*)(ws + OFF_W1H);
    float*    u1  = (float*)   (ws + OFF_U1);
    _Float16* s1  = (_Float16*)(ws + OFF_S1);
    float*    u2  = (float*)   (ws + OFF_U2);
    float*    out = (float*)d_out;             // (64, 500, 2) f32

    hipLaunchKernelGGL(w1_to_f16,  dim3((FH * KPAD + 255) / 256), dim3(256), 0, stream, W1, W1h);
    hipLaunchKernelGGL(gemm1_wmma, dim3(M_ / 16),                 dim3(256), 0, stream, x, W1h, u1);
    hipLaunchKernelGGL(scan1,      dim3((B_ * FH) / 256),         dim3(256), 0, stream, u1, s1);
    hipLaunchKernelGGL(layer2_mm,  dim3((M_ + 255) / 256),        dim3(256), 0, stream, s1, W2, u2);
    hipLaunchKernelGGL(scan2,      dim3(1),                       dim3(128), 0, stream, u2, out);
}
